// GroupedQueryAttention_31078383354636
// MI455X (gfx1250) — matrix operand order repaired and checked by reading, not compiled, not run on hardware
//
#include <hip/hip_runtime.h>

typedef __attribute__((ext_vector_type(16))) _Float16 v16h;
typedef __attribute__((ext_vector_type(8)))  _Float16 v8h;
typedef __attribute__((ext_vector_type(8)))  float    v8f;

// Problem constants
constexpr int BB   = 2;
constexpr int SLEN = 4096;
constexpr int DIM  = 2048;
constexpr int NH   = 16;
constexpr int NG   = 4;
constexpr int HD   = 128;
constexpr int WW   = 256;
constexpr int NB   = SLEN / WW;       // 16
constexpr int REP  = NH / NG;         // 4
constexpr float SCALE = 0.08838834764831845f; // HD^-0.5
constexpr float EPS = 1e-5f;

// Async DMA: global -> LDS, 16B per lane, tracked by ASYNCcnt (CDNA5 ISA 10.7)
__device__ __forceinline__ void async_b128(const void* sbase, unsigned voff,
                                           void* ldsPtr) {
  unsigned lds = (unsigned)(unsigned long long)ldsPtr; // low 32b of generic = LDS offset
  asm volatile("global_load_async_to_lds_b128 %0, %1, %2"
               :: "v"(lds), "v"(voff), "s"(sbase) : "memory");
}
__device__ __forceinline__ void async_wait0() {
  asm volatile("s_wait_asynccnt 0x0" ::: "memory");
}

// ---------------------------------------------------------------------------
// elementwise f32 -> f16
__global__ void k_f32_to_f16(const float* __restrict__ in,
                             _Float16* __restrict__ out, long n) {
  long i = (long)blockIdx.x * blockDim.x + threadIdx.x;
  long stride = (long)gridDim.x * blockDim.x;
  for (; i < n; i += stride) out[i] = (_Float16)in[i];
}

// w [K][N] f32  ->  wt [N][K] f16
__global__ void k_transpose_f16(const float* __restrict__ in,
                                _Float16* __restrict__ out, int K, int N) {
  long total = (long)K * N;
  long stride = (long)gridDim.x * blockDim.x;
  for (long i = (long)blockIdx.x * blockDim.x + threadIdx.x; i < total; i += stride) {
    long nn = i / K, kk = i % K;
    out[i] = (_Float16)in[kk * (long)N + nn];
  }
}

// ---------------------------------------------------------------------------
// C[M,N] = A[M,K] x B[K,N] (+bias), A f16 row-major, Bt = B^T f16 [N][K]
// 128x128 tile, BK=64 (2 WMMA k-steps per barrier), 8 waves -> 2x4 16x16 tiles.
// Tiles staged with GLOBAL_LOAD_ASYNC_TO_LDS_B128 (no VGPR round trip).
template <bool HAS_BIAS>
__global__ __launch_bounds__(256) void k_gemm_f16(
    const _Float16* __restrict__ A, const _Float16* __restrict__ Bt,
    float* __restrict__ C, const float* __restrict__ bias,
    int M, int N, int K) {
  constexpr int BK  = 64;
  constexpr int LDT = BK + 8;                   // 72 halves, 144B rows (16B aligned)
  __shared__ _Float16 As[128 * LDT];
  __shared__ _Float16 Bs[128 * LDT];

  const int tid  = threadIdx.x;
  const int bm   = blockIdx.y * 128;
  const int bn   = blockIdx.x * 128;
  const int wave = tid >> 5, lane = tid & 31;
  const int n16  = lane & 15, hi = lane >> 4;
  const int wm   = (wave >> 1) * 32;            // 0,32,64,96
  const int wn   = (wave & 1) * 64;             // 0,64

  v8f zero = {};
  v8f acc[2][4];
#pragma unroll
  for (int i = 0; i < 2; i++)
#pragma unroll
    for (int j = 0; j < 4; j++) acc[i][j] = zero;

  for (int kk = 0; kk < K; kk += BK) {
    // prefetch next K tile into cache while this one is staged/consumed
    if (kk + BK < K) {
      int pr = tid >> 1, pc = (tid & 1) * 32;
      __builtin_prefetch(&A[(long)(bm + pr) * K + kk + BK + pc], 0, 0);
      __builtin_prefetch(&Bt[(long)(bn + pr) * K + kk + BK + pc], 0, 0);
    }
#pragma unroll
    for (int it = 0; it < 4; it++) {
      int cid = tid + it * 256;                 // 0..1023
      int r = cid >> 3;
      int c = (cid & 7) * 8;
      unsigned aoff = (unsigned)(((long)(bm + r) * K + kk + c) * 2);
      unsigned boff = (unsigned)(((long)(bn + r) * K + kk + c) * 2);
      async_b128(A,  aoff, &As[r * LDT + c]);
      async_b128(Bt, boff, &Bs[r * LDT + c]);
    }
    async_wait0();
    __syncthreads();

#pragma unroll
    for (int ks = 0; ks < 2; ks++) {
      v16h afrag[2], bfrag[4];
#pragma unroll
      for (int mt = 0; mt < 2; mt++) {
        const _Float16* p = &As[(wm + mt * 16 + n16) * LDT + ks * 32];
        v8h lo = *(const v8h*)(p + hi * 16);
        v8h hh = *(const v8h*)(p + hi * 16 + 8);
#pragma unroll
        for (int i = 0; i < 8; i++) { afrag[mt][i] = lo[i]; afrag[mt][i + 8] = hh[i]; }
      }
#pragma unroll
      for (int nt = 0; nt < 4; nt++) {
        const _Float16* p = &Bs[(wn + nt * 16 + n16) * LDT + ks * 32 + hi * 16];
        v8h lo = *(const v8h*)p;
        v8h hh = *(const v8h*)(p + 8);
#pragma unroll
        for (int i = 0; i < 8; i++) { bfrag[nt][i] = lo[i]; bfrag[nt][i + 8] = hh[i]; }
      }
#pragma unroll
      for (int mt = 0; mt < 2; mt++)
#pragma unroll
        for (int nt = 0; nt < 4; nt++)
          acc[mt][nt] = __builtin_amdgcn_wmma_f32_16x16x32_f16(
              false, afrag[mt], false, bfrag[nt], (short)0, acc[mt][nt], false, false);
    }
    __syncthreads();
  }

#pragma unroll
  for (int nt = 0; nt < 4; nt++) {
    const int col = bn + wn + nt * 16 + n16;
    const float bv = HAS_BIAS ? bias[col] : 0.0f;
#pragma unroll
    for (int mt = 0; mt < 2; mt++) {
      const long base = (long)(bm + wm + mt * 16 + hi * 8) * N + col;
#pragma unroll
      for (int r = 0; r < 8; r++)
        C[base + (long)r * N] = acc[mt][nt][r] + bv;
    }
  }
}

// ---------------------------------------------------------------------------
// RoPE + LayerNorm over HD=128, one wave per (row, head) vector, write f16.
__global__ __launch_bounds__(256) void k_ropeln(
    const float* __restrict__ in, _Float16* __restrict__ out,
    const float* __restrict__ cosb, const float* __restrict__ sinb,
    const float* __restrict__ scale, const float* __restrict__ shift,
    int nHeads, long nVec) {
  long vec = (long)blockIdx.x * 8 + (threadIdx.x >> 5);
  int lane = threadIdx.x & 31;
  if (vec >= nVec) return;
  int hIdx = (int)(vec % nHeads);
  long row = vec / nHeads;                 // b*S + s
  int s = (int)(row % SLEN);
  const float* base = in + row * (long)(nHeads * HD) + hIdx * HD;

  float t[4], r[4];
  int hd[4];
#pragma unroll
  for (int j = 0; j < 4; j++) { hd[j] = lane + 32 * j; t[j] = base[hd[j]]; }
#pragma unroll
  for (int j = 0; j < 4; j++) {
    float rot = (j < 2) ? -t[j + 2] : t[j - 2];
    r[j] = t[j] * cosb[s * HD + hd[j]] + rot * sinb[s * HD + hd[j]];
  }
  float sum = r[0] + r[1] + r[2] + r[3];
#pragma unroll
  for (int m = 16; m >= 1; m >>= 1) sum += __shfl_xor(sum, m, 32);
  float mean = sum * (1.0f / HD);
  float var = 0.f;
#pragma unroll
  for (int j = 0; j < 4; j++) { float d = r[j] - mean; var += d * d; }
#pragma unroll
  for (int m = 16; m >= 1; m >>= 1) var += __shfl_xor(var, m, 32);
  float inv = 1.0f / (sqrtf(var * (1.0f / HD)) + EPS);

  _Float16* ob = out + row * (long)(nHeads * HD) + hIdx * HD;
#pragma unroll
  for (int j = 0; j < 4; j++)
    ob[hd[j]] = (_Float16)(scale[hd[j]] * (r[j] - mean) * inv + shift[hd[j]]);
}

// ---------------------------------------------------------------------------
// Sliding-window attention. Grid: B*H*NB*2 blocks; each handles 128 q rows of
// one window block against the 512-key window in 4 chunks of 128 keys with
// online (flash) softmax. All GEMMs via v_wmma_f32_16x16x32_f16.
__global__ __launch_bounds__(256) void k_attn(
    const _Float16* __restrict__ qh, const _Float16* __restrict__ kh,
    const _Float16* __restrict__ vh, _Float16* __restrict__ ctx) {
  extern __shared__ _Float16 smem[];
  constexpr int LDQ = HD + 8;               // 136 halves, 272B rows (16B aligned)
  _Float16* Qs = smem;                      // 128 x LDQ   (q rows x hd)
  _Float16* Ks = Qs + 128 * LDQ;            // 128 x LDQ   (keys  x hd)
  _Float16* Vt = Ks + 128 * LDQ;            // 128 x LDQ   (hd    x keys)
  _Float16* Ps = Vt + 128 * LDQ;            // 8 waves x 16 x LDQ

  const int idx = blockIdx.x;
  const int sub = idx & 1;
  const int blk = (idx >> 1) & (NB - 1);
  const int h   = (idx >> 5) & (NH - 1);
  const int b   = idx >> 9;
  const int g   = h / REP;
  const int tid = threadIdx.x, wave = tid >> 5, lane = tid & 31;
  const int n16 = lane & 15, hi = lane >> 4;
  const int qBase = blk * WW + sub * 128;
  const long rowB = (long)b * SLEN;
  const int m0 = wave * 16;

  // stage Q tile via async DMA
#pragma unroll
  for (int it = 0; it < 8; it++) {
    int cid = tid + it * 256;               // 0..2047
    int r = cid >> 4;
    int c = (cid & 15) * 8;
    unsigned goff = (unsigned)((((rowB + qBase + r) * (long)DIM) + h * HD + c) * 2);
    async_b128(qh, goff, &Qs[r * LDQ + c]);
  }

  v8f zero = {};
  v8f cacc[8];
#pragma unroll
  for (int i = 0; i < 8; i++) cacc[i] = zero;
  float mrun[8], lrun[8];
#pragma unroll
  for (int r = 0; r < 8; r++) { mrun[r] = -__builtin_inff(); lrun[r] = 0.f; }
  async_wait0();
  __syncthreads();

  for (int ch = 0; ch < 4; ch++) {
    if (blk == 0 && ch < 2) continue;       // entire chunk precedes sequence
    const int kBase = (blk - 1) * WW + ch * 128;   // >= 0 here

    // stage K tile [key][hd] (async DMA) and V transposed [hd][key] (scatter)
#pragma unroll
    for (int it = 0; it < 8; it++) {
      int cid = tid + it * 256;
      int kr = cid >> 4;
      int c0 = (cid & 15) * 8;
      const long krow = (rowB + kBase + kr) * 512 + g * HD + c0;
      async_b128(kh, (unsigned)(krow * 2), &Ks[kr * LDQ + c0]);
      v8h vv = *(const v8h*)&vh[krow];
#pragma unroll
      for (int j = 0; j < 8; j++) Vt[(c0 + j) * LDQ + kr] = vv[j];
    }
    async_wait0();
    __syncthreads();

    // scores: S[16 q][128 k] per wave, 8 k-tiles, K-dim = HD in 4 steps of 32
    v8f sacc[8];
#pragma unroll
    for (int i = 0; i < 8; i++) sacc[i] = zero;
    for (int ks = 0; ks < 4; ks++) {
      v16h af;
      {
        const _Float16* p = &Qs[(m0 + n16) * LDQ + ks * 32];
        v8h lo = *(const v8h*)(p + hi * 16);
        v8h hh = *(const v8h*)(p + hi * 16 + 8);
#pragma unroll
        for (int i = 0; i < 8; i++) { af[i] = lo[i]; af[i + 8] = hh[i]; }
      }
#pragma unroll
      for (int kt = 0; kt < 8; kt++) {
        v16h bf;
        const _Float16* p = &Ks[(kt * 16 + n16) * LDQ + ks * 32 + hi * 16];
        v8h lo = *(const v8h*)p;
        v8h hh = *(const v8h*)(p + 8);
#pragma unroll
        for (int i = 0; i < 8; i++) { bf[i] = lo[i]; bf[i + 8] = hh[i]; }
        sacc[kt] = __builtin_amdgcn_wmma_f32_16x16x32_f16(
            false, af, false, bf, (short)0, sacc[kt], false, false);
      }
    }

    // band mask + per-row max (rows live in 16-lane halves of the wave)
    float rowmax[8];
#pragma unroll
    for (int r = 0; r < 8; r++) rowmax[r] = -__builtin_inff();
#pragma unroll
    for (int kt = 0; kt < 8; kt++)
#pragma unroll
      for (int r = 0; r < 8; r++) {
        int qg = qBase + m0 + 8 * hi + r;
        int kg = kBase + kt * 16 + n16;
        float s = sacc[kt][r] * SCALE;
        bool ok = (kg <= qg) && (qg - kg < WW);
        s = ok ? s : -__builtin_inff();
        sacc[kt][r] = s;
        rowmax[r] = fmaxf(rowmax[r], s);
      }
#pragma unroll
    for (int r = 0; r < 8; r++)
#pragma unroll
      for (int m = 8; m >= 1; m >>= 1)
        rowmax[r] = fmaxf(rowmax[r], __shfl_xor(rowmax[r], m, 32));

    float corr[8], rowsum[8];
#pragma unroll
    for (int r = 0; r < 8; r++) {
      float mn = fmaxf(mrun[r], rowmax[r]);
      corr[r] = (mn == -__builtin_inff()) ? 1.f : __expf(mrun[r] - mn);
      mrun[r] = mn;
      rowsum[r] = 0.f;
    }

    _Float16* Pw = Ps + wave * 16 * LDQ;
#pragma unroll
    for (int kt = 0; kt < 8; kt++)
#pragma unroll
      for (int r = 0; r < 8; r++) {
        float mn = mrun[r];
        float p = (mn == -__builtin_inff()) ? 0.f : __expf(sacc[kt][r] - mn);
        rowsum[r] += p;
        Pw[(8 * hi + r) * LDQ + kt * 16 + n16] = (_Float16)p;
      }
#pragma unroll
    for (int r = 0; r < 8; r++) {
#pragma unroll
      for (int m = 8; m >= 1; m >>= 1) rowsum[r] += __shfl_xor(rowsum[r], m, 32);
      lrun[r] = lrun[r] * corr[r] + rowsum[r];
#pragma unroll
      for (int nt = 0; nt < 8; nt++) cacc[nt][r] *= corr[r];
    }

    // ctx += P[16,128] x V[128,128]; K-dim = keys in 4 steps of 32
    for (int ks = 0; ks < 4; ks++) {
      v16h af;
      {
        const _Float16* p = &Pw[n16 * LDQ + ks * 32];
        v8h lo = *(const v8h*)(p + hi * 16);
        v8h hh = *(const v8h*)(p + hi * 16 + 8);
#pragma unroll
        for (int i = 0; i < 8; i++) { af[i] = lo[i]; af[i + 8] = hh[i]; }
      }
#pragma unroll
      for (int nt = 0; nt < 8; nt++) {
        v16h bf;
        const _Float16* p = &Vt[(nt * 16 + n16) * LDQ + ks * 32 + hi * 16];
        v8h lo = *(const v8h*)p;
        v8h hh = *(const v8h*)(p + 8);
#pragma unroll
        for (int i = 0; i < 8; i++) { bf[i] = lo[i]; bf[i + 8] = hh[i]; }
        cacc[nt] = __builtin_amdgcn_wmma_f32_16x16x32_f16(
            false, af, false, bf, (short)0, cacc[nt], false, false);
      }
    }
    __syncthreads();
  }

  // normalize and write ctx as f16 [b*s][h*HD]
#pragma unroll
  for (int r = 0; r < 8; r++) {
    float inv = 1.0f / lrun[r];
    int qg = qBase + m0 + 8 * hi + r;
#pragma unroll
    for (int nt = 0; nt < 8; nt++)
      ctx[(rowB + qg) * (long)DIM + h * HD + nt * 16 + n16] =
          (_Float16)(cacc[nt][r] * inv);
  }
}

// ---------------------------------------------------------------------------
extern "C" void kernel_launch(void* const* d_in, const int* in_sizes, int n_in,
                              void* d_out, int out_size, void* d_ws, size_t ws_size,
                              hipStream_t stream) {
  (void)in_sizes; (void)n_in; (void)out_size; (void)ws_size;
  const float* x     = (const float*)d_in[0];
  const float* cosb  = (const float*)d_in[2];
  const float* sinb  = (const float*)d_in[3];
  const float* wq    = (const float*)d_in[5];
  const float* wk    = (const float*)d_in[6];
  const float* wv    = (const float*)d_in[7];
  const float* wo    = (const float*)d_in[8];
  const float* bo    = (const float*)d_in[9];
  const float* q_sc  = (const float*)d_in[10];
  const float* q_sh  = (const float*)d_in[11];
  const float* k_sc  = (const float*)d_in[12];
  const float* k_sh  = (const float*)d_in[13];

  char* ws = (char*)d_ws;
  size_t off = 0;
  auto alloc = [&](size_t bytes) -> void* {
    void* p = ws + off;
    off += (bytes + 255) & ~(size_t)255;
    return p;
  };
  const long M = (long)BB * SLEN;                        // 8192 rows

  _Float16* xh   = (_Float16*)alloc(M * DIM * 2);
  _Float16* wqt  = (_Float16*)alloc((long)DIM * DIM * 2);
  _Float16* wkt  = (_Float16*)alloc(512L * DIM * 2);
  _Float16* wvt  = (_Float16*)alloc(512L * DIM * 2);
  _Float16* wot  = (_Float16*)alloc((long)DIM * DIM * 2);
  float*    qf   = (float*)alloc(M * DIM * 4);
  float*    kf   = (float*)alloc(M * 512 * 4);
  float*    vf   = (float*)alloc(M * 512 * 4);
  _Float16* qhh  = (_Float16*)alloc(M * DIM * 2);
  _Float16* khh  = (_Float16*)alloc(M * 512 * 2);
  _Float16* vhh  = (_Float16*)alloc(M * 512 * 2);
  _Float16* ctxh = (_Float16*)alloc(M * DIM * 2);

  k_f32_to_f16<<<4096, 256, 0, stream>>>(x, xh, M * DIM);
  k_transpose_f16<<<4096, 256, 0, stream>>>(wq, wqt, DIM, DIM);
  k_transpose_f16<<<2048, 256, 0, stream>>>(wk, wkt, DIM, 512);
  k_transpose_f16<<<2048, 256, 0, stream>>>(wv, wvt, DIM, 512);
  k_transpose_f16<<<4096, 256, 0, stream>>>(wo, wot, DIM, DIM);

  k_gemm_f16<false><<<dim3(DIM / 128, M / 128), 256, 0, stream>>>(xh, wqt, qf, nullptr, (int)M, DIM, DIM);
  k_gemm_f16<false><<<dim3(512 / 128, M / 128), 256, 0, stream>>>(xh, wkt, kf, nullptr, (int)M, 512, DIM);
  k_gemm_f16<false><<<dim3(512 / 128, M / 128), 256, 0, stream>>>(xh, wvt, vf, nullptr, (int)M, 512, DIM);

  k_ropeln<<<(unsigned)(M * NH / 8), 256, 0, stream>>>(qf, qhh, cosb, sinb, q_sc, q_sh, NH, M * NH);
  k_ropeln<<<(unsigned)(M * NG / 8), 256, 0, stream>>>(kf, khh, cosb, sinb, k_sc, k_sh, NG, M * NG);
  k_f32_to_f16<<<2048, 256, 0, stream>>>(vf, vhh, M * 512);

  const size_t smem = (size_t)(3 * 128 + 8 * 16) * (HD + 8) * sizeof(_Float16); // ~136KB
  k_attn<<<BB * NH * NB * 2, 256, smem, stream>>>(qhh, khh, vhh, ctxh);

  k_gemm_f16<true><<<dim3(DIM / 128, M / 128), 256, 0, stream>>>(ctxh, wot, (float*)d_out, bo, (int)M, DIM, DIM);
}
